// TipAdapterHead_26414048870750
// MI455X (gfx1250) — matrix operand order repaired and checked by reading, not compiled, not run on hardware
//
#include <hip/hip_runtime.h>
#include <hip/hip_bf16.h>

typedef __attribute__((ext_vector_type(16))) _Float16 v16h;
typedef __attribute__((ext_vector_type(8)))  _Float16 v8h;
typedef __attribute__((ext_vector_type(8)))  float    v8f;

#define L_   4
#define B_   16
#define N_   197
#define NP_  208        // 13 tiles of 16
#define NT_  13
#define D_   768
#define S_   4096
#define BETA 6.8f

// ---------------------------------------------------------------------------
// Kernel 1: L2-normalize cache keys -> f16 row-major kn[L,S,D] and f16
// transposed knT[L,D,S].  One wave per key row (D=768 -> 24 elems/lane).
// ---------------------------------------------------------------------------
__global__ __launch_bounds__(256) void norm_keys(const float* __restrict__ ck,
                                                 _Float16* __restrict__ kn,
                                                 _Float16* __restrict__ knT) {
  const int row  = blockIdx.x * 8 + (threadIdx.x >> 5);   // 0 .. L*S-1
  const int lane = threadIdx.x & 31;
  const int l = row >> 12, s = row & (S_ - 1);

  const float* src = ck + (size_t)row * D_;
  float x[24];
  float ss = 0.f;
#pragma unroll
  for (int i = 0; i < 24; ++i) { x[i] = src[lane + i * 32]; ss += x[i] * x[i]; }
#pragma unroll
  for (int off = 16; off > 0; off >>= 1) ss += __shfl_xor(ss, off, 32);
  const float inv = 1.0f / fmaxf(sqrtf(ss), 1e-12f);

  _Float16* dst  = kn  + (size_t)row * D_;
  _Float16* dstT = knT + (size_t)l * D_ * S_ + s;
#pragma unroll
  for (int i = 0; i < 24; ++i) {
    const int d = lane + i * 32;
    const _Float16 v = (_Float16)(x[i] * inv);
    dst[d] = v;
    dstT[(size_t)d * S_] = v;
  }
}

// ---------------------------------------------------------------------------
// Kernel 2: normalize queries -> f16 qn[L,B,208,D], rows >= 197 zero-padded.
// ---------------------------------------------------------------------------
__global__ __launch_bounds__(256) void norm_q(const float* __restrict__ q,
                                              _Float16* __restrict__ qn) {
  const int row  = blockIdx.x * 8 + (threadIdx.x >> 5);   // 0 .. L*B*208-1
  const int lane = threadIdx.x & 31;
  const int n  = row % NP_;
  const int lb = row / NP_;

  _Float16* dst = qn + (size_t)row * D_;
  if (n >= N_) {
#pragma unroll
    for (int i = 0; i < 24; ++i) dst[lane + i * 32] = (_Float16)0.f;
    return;
  }
  const float* src = q + ((size_t)lb * N_ + n) * D_;
  float x[24];
  float ss = 0.f;
#pragma unroll
  for (int i = 0; i < 24; ++i) { x[i] = src[lane + i * 32]; ss += x[i] * x[i]; }
#pragma unroll
  for (int off = 16; off > 0; off >>= 1) ss += __shfl_xor(ss, off, 32);
  const float inv = 1.0f / fmaxf(sqrtf(ss), 1e-12f);
#pragma unroll
  for (int i = 0; i < 24; ++i) dst[lane + i * 32] = (_Float16)(x[i] * inv);
}

// ---------------------------------------------------------------------------
// Kernel 3: main.  Grid (13, 16, 4) = (ntile, b, l); 256 threads = 8 waves.
// GEMM1 (WMMA f16) -> scores in 128KB LDS -> softmax -> GEMM2 (WMMA f16)
// -> f32 partial[L,B,13,16,768].
// ---------------------------------------------------------------------------
__global__ __launch_bounds__(256) void tip_main(const _Float16* __restrict__ qn,
                                                const _Float16* __restrict__ kn,
                                                const _Float16* __restrict__ knT,
                                                float* __restrict__ partial) {
  __shared__ _Float16 scores[16 * S_];   // 131072 bytes

  const int ntile = blockIdx.x, b = blockIdx.y, l = blockIdx.z;
  const int lane = threadIdx.x & 31;
  const int wid  = threadIdx.x >> 5;
  const int hi   = lane >> 4;
  const int lm   = lane & 15;

  // ---------------- GEMM1: scores[16 x 512-chunk] = BETA * q @ k^T ---------
  // A layout (f16 16x32): lane m=lm, half j -> K = hi*16 + j
  // B layout (f16 32x16): lane n=lm, half j -> K = hi*16 + j
  const _Float16* qrow =
      qn + ((size_t)(l * B_ + b) * NP_ + ntile * 16 + lm) * D_;

  for (int kt = 0; kt < 32; ++kt) {
    const int key0 = wid * 512 + kt * 16;
    const _Float16* krow = kn + ((size_t)l * S_ + key0 + lm) * D_;
    v8f acc = {};
#pragma unroll
    for (int dk = 0; dk < 24; ++dk) {
      const int kb = dk * 32;
      v16h a, bf;
      *(v8h*)&a         = *(const v8h*)(qrow + kb + hi * 16);       // j=0..7
      *((v8h*)&a + 1)   = *(const v8h*)(qrow + kb + hi * 16 + 8);   // j=8..15
      bf                = *(const v16h*)(krow + kb + hi * 16);      // j=0..15
      acc = __builtin_amdgcn_wmma_f32_16x16x32_f16(
          false, a, false, bf, (short)0, acc, false, false);
    }
    // C layout: lane n = lm (key col), VGPR v -> row m = v + hi*8
#pragma unroll
    for (int v = 0; v < 8; ++v)
      scores[(v + hi * 8) * S_ + key0 + lm] = (_Float16)(acc[v] * BETA);
  }
  __syncthreads();

  // ---------------- softmax over S, in place (f16 weights) -----------------
#pragma unroll
  for (int rr = 0; rr < 2; ++rr) {
    const int r = wid * 2 + rr;
    _Float16* row = scores + r * S_;
    float mx = -1e30f;
    for (int i = lane; i < S_; i += 32) mx = fmaxf(mx, (float)row[i]);
#pragma unroll
    for (int off = 16; off > 0; off >>= 1)
      mx = fmaxf(mx, __shfl_xor(mx, off, 32));
    float sum = 0.f;
    for (int i = lane; i < S_; i += 32) sum += __expf((float)row[i] - mx);
#pragma unroll
    for (int off = 16; off > 0; off >>= 1) sum += __shfl_xor(sum, off, 32);
    const float inv = 1.0f / sum;
    for (int i = lane; i < S_; i += 32)
      row[i] = (_Float16)(__expf((float)row[i] - mx) * inv);
  }
  __syncthreads();

  // ---------------- GEMM2: partial[16 x 96-chunk] = W @ k_norm -------------
  v8f acc2[6] = {};
  const int d0w = wid * 96;
  for (int st = 0; st < 128; ++st) {
    const int sb = st * 32;
    v16h a;
    const _Float16* arow = scores + lm * S_ + sb + hi * 16;
    *(v8h*)&a       = *(const v8h*)(arow);
    *((v8h*)&a + 1) = *(const v8h*)(arow + 8);
#pragma unroll
    for (int t = 0; t < 6; ++t) {
      const _Float16* brow =
          knT + ((size_t)l * D_ + d0w + t * 16 + lm) * S_ + sb + hi * 16;
      const v16h bf = *(const v16h*)brow;
      acc2[t] = __builtin_amdgcn_wmma_f32_16x16x32_f16(
          false, a, false, bf, (short)0, acc2[t], false, false);
    }
  }
  float* pdst =
      partial + ((((size_t)l * B_ + b) * NT_ + ntile) * 16) * (size_t)D_;
#pragma unroll
  for (int t = 0; t < 6; ++t)
#pragma unroll
    for (int v = 0; v < 8; ++v)
      pdst[(size_t)(v + hi * 8) * D_ + d0w + t * 16 + lm] = acc2[t][v];
}

// ---------------------------------------------------------------------------
// Kernel 4: sum partials over L -> out[B,197,D] (deterministic, no atomics)
// ---------------------------------------------------------------------------
__global__ __launch_bounds__(256) void reduce_l(const float* __restrict__ partial,
                                                float* __restrict__ out) {
  const size_t idx = (size_t)blockIdx.x * blockDim.x + threadIdx.x;
  const size_t total = (size_t)B_ * N_ * D_;
  if (idx >= total) return;
  const int d = (int)(idx % D_);
  const size_t t = idx / D_;
  const int n = (int)(t % N_);
  const int b = (int)(t / N_);
  const int nt = n >> 4, r = n & 15;
  float s = 0.f;
#pragma unroll
  for (int l = 0; l < L_; ++l)
    s += partial[((((size_t)l * B_ + b) * NT_ + nt) * 16 + r) * (size_t)D_ + d];
  out[idx] = s;
}

// ---------------------------------------------------------------------------
extern "C" void kernel_launch(void* const* d_in, const int* in_sizes, int n_in,
                              void* d_out, int out_size, void* d_ws, size_t ws_size,
                              hipStream_t stream) {
  const float* queries    = (const float*)d_in[0];   // (L,B,N,D) f32
  const float* cache_keys = (const float*)d_in[1];   // (L,S,D)   f32
  float* out = (float*)d_out;                        // (B,N,D)   f32

  char* ws = (char*)d_ws;
  size_t off = 0;
  _Float16* qn  = (_Float16*)(ws + off); off += (size_t)L_ * B_ * NP_ * D_ * 2;  // 19.5 MB
  _Float16* kn  = (_Float16*)(ws + off); off += (size_t)L_ * S_ * D_ * 2;        // 24 MB
  _Float16* knT = (_Float16*)(ws + off); off += (size_t)L_ * D_ * S_ * 2;        // 24 MB
  float* partial = (float*)(ws + off);   off += (size_t)L_ * B_ * NT_ * 16 * D_ * 4; // 39 MB

  // 1) normalize keys (row-major f16 + transposed f16)
  norm_keys<<<(L_ * S_) / 8, 256, 0, stream>>>(cache_keys, kn, knT);
  // 2) normalize + pad queries
  norm_q<<<(L_ * B_ * NP_) / 8, 256, 0, stream>>>(queries, qn);
  // 3) main fused attention kernel
  tip_main<<<dim3(NT_, B_, L_), 256, 0, stream>>>(qn, kn, knT, partial);
  // 4) reduce over layers
  const size_t total = (size_t)B_ * N_ * D_;
  reduce_l<<<(unsigned)((total + 255) / 256), 256, 0, stream>>>(partial, out);
}
